// MixHopNetwork_48541720379665
// MI455X (gfx1250) — matrix operand order repaired and checked by reading, not compiled, not run on hardware
//
#include <hip/hip_runtime.h>

#define FEAT 512
#define HID  200
#define CLS  64

typedef __attribute__((ext_vector_type(16))) __bf16 bf16x16;
typedef __attribute__((ext_vector_type(8)))  __bf16 bf16x8;
typedef __attribute__((ext_vector_type(8)))  float  f32x8;

union bfrag {
  bf16x16 v;
  struct { bf16x8 lo, hi; } h;
};

__device__ __forceinline__ unsigned pack2bf(float a, float b) {
  __bf16 ba = (__bf16)a;                 // fptrunc, RNE, native bf16 cvt
  __bf16 bb = (__bf16)b;
  unsigned short lo, hi;
  __builtin_memcpy(&lo, &ba, 2);
  __builtin_memcpy(&hi, &bb, 2);
  return (unsigned)lo | ((unsigned)hi << 16);
}

// Tiled bf16-WMMA GEMM: C[M x Ncols] = A[M x K] @ W (+bias/relu).
// Block: 256 threads = 8 waves, tile 128(M) x 64(N), K-step 32.
// seg==1: W is [3, K, HID]; output column c belongs to MixHop block c/HID.
// mode: 0 = +bias,relu ; 1 = none ; 2 = +bias.
// Column blocks routed to (dst0,sd0)/(dst1,sd1)/(dst2,sd2).
__global__ __launch_bounds__(256)
void mixhop_wmma_gemm(const float* __restrict__ A, int lda, int M, int K, int Ncols,
                      const float* __restrict__ W, int seg,
                      const float* __restrict__ bias, int mode,
                      float* __restrict__ dst0, int sd0,
                      float* __restrict__ dst1, int sd1,
                      float* __restrict__ dst2, int sd2)
{
  // Row stride 40 bf16 = 80 bytes: 16B-aligned rows, staggered banks.
  __shared__ __align__(16) __bf16 lds_a[128][40];  // A tile, row-major [m][k]
  __shared__ __align__(16) __bf16 lds_bt[64][40];  // B tile, TRANSPOSED [n][k]

  const int tid   = threadIdx.x;
  const int lane  = tid & 31;
  const int wv    = tid >> 5;            // 8 waves
  const int mBase = blockIdx.x * 128;
  const int nBase = blockIdx.y * 64;
  const int half  = lane >> 4;           // 0 | 1 (lane half)

  const bool fullM = (mBase + 128 <= M);
  const bool fullN = (nBase + 64 <= Ncols);

  f32x8 acc[4] = {};

  for (int k0 = 0; k0 < K; k0 += 32) {
    const bool fullK = (k0 + 32 <= K);

    // ---- stage A tile: 128x32 = 2048 bf16-pairs, 8 pairs/thread ----
    if (fullM && fullK) {                // interior tile: unguarded b64 loads
      #pragma unroll
      for (int t = 0; t < 8; ++t) {
        int idx = tid + 256 * t;
        int r  = idx >> 4;               // 0..127
        int kp = idx & 15;               // 0..15 (pair of k)
        const float2 a2 = *(const float2*)&A[(long)(mBase + r) * lda + k0 + 2 * kp];
        *(unsigned*)&lds_a[r][2 * kp] = pack2bf(a2.x, a2.y);
      }
    } else {                             // boundary tile: guarded scalar loads
      #pragma unroll
      for (int t = 0; t < 8; ++t) {
        int idx = tid + 256 * t;
        int r  = idx >> 4;
        int kp = idx & 15;
        int gm = mBase + r, gk = k0 + 2 * kp;
        float a0 = (gm < M && gk     < K) ? A[(long)gm * lda + gk]     : 0.f;
        float a1 = (gm < M && gk + 1 < K) ? A[(long)gm * lda + gk + 1] : 0.f;
        *(unsigned*)&lds_a[r][2 * kp] = pack2bf(a0, a1);
      }
    }

    // ---- stage B tile transposed: 64x32 = 1024 pairs, 4 pairs/thread ----
    if (fullN && fullK) {
      #pragma unroll
      for (int t = 0; t < 4; ++t) {
        int idx = tid + 256 * t;
        int c  = idx >> 4;               // 0..63 (output column)
        int kp = idx & 15;
        int gk = k0 + 2 * kp, gc = nBase + c;
        float w0, w1;
        if (seg) {
          int blk = gc / HID, lc = gc - blk * HID;
          const float* wb = W + (long)blk * K * HID + lc;
          w0 = wb[(long)gk * HID];
          w1 = wb[(long)(gk + 1) * HID];
        } else {
          w0 = W[(long)gk * Ncols + gc];
          w1 = W[(long)(gk + 1) * Ncols + gc];
        }
        *(unsigned*)&lds_bt[c][2 * kp] = pack2bf(w0, w1);
      }
    } else {
      #pragma unroll
      for (int t = 0; t < 4; ++t) {
        int idx = tid + 256 * t;
        int c  = idx >> 4;
        int kp = idx & 15;
        int gk = k0 + 2 * kp, gc = nBase + c;
        float w0 = 0.f, w1 = 0.f;
        if (gc < Ncols) {
          if (seg) {
            int blk = gc / HID, lc = gc - blk * HID;
            const float* wb = W + (long)blk * K * HID + lc;
            if (gk     < K) w0 = wb[(long)gk * HID];
            if (gk + 1 < K) w1 = wb[(long)(gk + 1) * HID];
          } else {
            if (gk     < K) w0 = W[(long)gk * Ncols + gc];
            if (gk + 1 < K) w1 = W[(long)(gk + 1) * Ncols + gc];
          }
        }
        *(unsigned*)&lds_bt[c][2 * kp] = pack2bf(w0, w1);
      }
    }
    __syncthreads();

    if (k0 + 32 < K) {   // speculative prefetch of next A tile -> global_prefetch_b8
      __builtin_prefetch(&A[(long)(mBase + (tid >> 1)) * lda + (k0 + 32)], 0, 1);
    }

    // ---- A fragment: two ds_load_b128 (K kb..kb+7 and 16+kb..16+kb+7) ----
    bfrag af;
    {
      const bf16x8* pa = (const bf16x8*)&lds_a[wv * 16 + (lane & 15)][0]; // 5 vec8/row
      af.h.lo = pa[half];        // K =      half*8 .. +7
      af.h.hi = pa[2 + half];    // K = 16 + half*8 .. +7
    }

    // ---- load all 4 B fragments first so DS loads overlap the WMMAs ----
    bfrag bf[4];
    #pragma unroll
    for (int t = 0; t < 4; ++t) {
      const bf16x8* pb = (const bf16x8*)&lds_bt[t * 16 + (lane & 15)][0];
      bf[t].h.lo = pb[half];
      bf[t].h.hi = pb[2 + half];
    }
    #pragma unroll
    for (int t = 0; t < 4; ++t) {
      acc[t] = __builtin_amdgcn_wmma_f32_16x16x32_bf16(
          false, af.v, false, bf[t].v, (short)0, acc[t], false, false);
    }
    __syncthreads();
  }

  // ---- epilogue: C layout VGPR r -> M = half*8 + r, N = lane&15 ----
  const int mrow = mBase + wv * 16 + half * 8;
  #pragma unroll
  for (int t = 0; t < 4; ++t) {
    int n = nBase + t * 16 + (lane & 15);
    #pragma unroll
    for (int r = 0; r < 8; ++r) {
      int m = mrow + r;
      if (m < M && n < Ncols) {
        float v = acc[t][r];
        if (mode == 0)      v = fmaxf(v + bias[n], 0.f);
        else if (mode == 2) v = v + bias[n];
        float* dp; int st, lc;
        if (Ncols == 3 * HID) {
          int blk = n / HID; lc = n - blk * HID;
          dp = (blk == 0) ? dst0 : (blk == 1 ? dst1 : dst2);
          st = (blk == 0) ? sd0  : (blk == 1 ? sd1  : sd2);
        } else { dp = dst0; st = sd0; lc = n; }
        dp[(long)m * st + lc] = v;
      }
    }
  }
}

// One wave per edge: y[row[e]] += vals[e] * x[col[e]]  (200-wide rows, strided)
__global__ __launch_bounds__(256)
void mixhop_spmm(const long long* __restrict__ rows, const long long* __restrict__ cols,
                 const float* __restrict__ vals, int nedges,
                 const float* __restrict__ x, int sx,
                 float* __restrict__ y, int sy)
{
  int e    = (blockIdx.x * blockDim.x + threadIdx.x) >> 5;
  int lane = threadIdx.x & 31;
  if (e >= nedges) return;
  long r = (long)rows[e];
  long c = (long)cols[e];
  float v = vals[e];
  const float* xs = x + c * sx;
  float*       yd = y + r * sy;
  #pragma unroll 2
  for (int f = lane; f < HID; f += 32)
    atomicAdd(&yd[f], v * xs[f]);
}

__global__ void mixhop_bias600(float* __restrict__ y, const float* __restrict__ b, long total)
{
  long i = (long)blockIdx.x * blockDim.x + threadIdx.x;
  if (i < total) y[i] += b[i % (3 * HID)];
}

// One wave per row of 64 logits: lanes hold cols {lane, lane+32}
__global__ __launch_bounds__(256)
void mixhop_logsoftmax(const float* __restrict__ z, float* __restrict__ out, int nrows)
{
  int row  = (blockIdx.x * blockDim.x + threadIdx.x) >> 5;
  int lane = threadIdx.x & 31;
  if (row >= nrows) return;
  float a = z[(long)row * CLS + lane];
  float b = z[(long)row * CLS + 32 + lane];
  float mx = fmaxf(a, b);
  #pragma unroll
  for (int o = 16; o > 0; o >>= 1) mx = fmaxf(mx, __shfl_xor(mx, o, 32));
  float s = __expf(a - mx) + __expf(b - mx);
  #pragma unroll
  for (int o = 16; o > 0; o >>= 1) s += __shfl_xor(s, o, 32);
  float l = mx + __logf(s);
  out[(long)row * CLS + lane]      = a - l;
  out[(long)row * CLS + 32 + lane] = b - l;
}

extern "C" void kernel_launch(void* const* d_in, const int* in_sizes, int n_in,
                              void* d_out, int out_size, void* d_ws, size_t ws_size,
                              hipStream_t stream)
{
  const long long* adj  = (const long long*)d_in[0];   // [2, E] int64
  const float*     vals = (const float*)d_in[1];       // [E]
  const float*     feat = (const float*)d_in[2];       // [N, 512]
  const float*     W1   = (const float*)d_in[3];       // [3, 512, 200]
  const float*     b1   = (const float*)d_in[4];       // [3, 200] -> flat 600
  const float*     W2   = (const float*)d_in[5];       // [3, 600, 200]
  const float*     b2   = (const float*)d_in[6];       // [3, 200] -> flat 600
  const float*     Wfc  = (const float*)d_in[7];       // [600, 64]
  const float*     bfc  = (const float*)d_in[8];       // [64]

  const int E = in_sizes[1];
  const int N = in_sizes[2] / FEAT;
  const long long* rowsI = adj;
  const long long* colsI = adj + E;

  // Workspace: A1 + A2 (N x 600 each) + two N x 200 temps => N*1600 floats (~640 MB)
  float* A1 = (float*)d_ws;
  float* A2 = A1 + (size_t)N * 3 * HID;
  float* T1 = A2 + (size_t)N * 3 * HID;
  float* T2 = T1 + (size_t)N * HID;

  float* node_emb = (float*)d_out;
  float* preds    = node_emb + (size_t)N * CLS;

  const dim3 blk(256);
  const int  mT = (N + 127) / 128;
  const dim3 g600(mT, (3 * HID + 63) / 64);   // 10 column tiles
  const dim3 g64(mT, 1);
  const int  spmmGrid = (E + 7) / 8;          // 8 edge-waves per 256-thread block

  // ================= Layer 1: relu(feat @ W1[i] + b1[i]), i hops =================
  hipMemsetAsync(A1, 0, (size_t)N * 3 * HID * sizeof(float), stream);
  mixhop_wmma_gemm<<<g600, blk, 0, stream>>>(feat, FEAT, N, FEAT, 3 * HID, W1, 1, b1, 0,
                                             A1, 3 * HID, T1, HID, T2, HID);
  mixhop_spmm<<<spmmGrid, 256, 0, stream>>>(rowsI, colsI, vals, E, T1, HID, A1 + HID, 3 * HID);
  hipMemsetAsync(T1, 0, (size_t)N * HID * sizeof(float), stream);
  mixhop_spmm<<<spmmGrid, 256, 0, stream>>>(rowsI, colsI, vals, E, T2, HID, T1, HID);
  mixhop_spmm<<<spmmGrid, 256, 0, stream>>>(rowsI, colsI, vals, E, T1, HID, A1 + 2 * HID, 3 * HID);

  // ================= Layer 2: (A1 @ W2[i]) hops, then + b2 =======================
  hipMemsetAsync(A2, 0, (size_t)N * 3 * HID * sizeof(float), stream);
  mixhop_wmma_gemm<<<g600, blk, 0, stream>>>(A1, 3 * HID, N, 3 * HID, 3 * HID, W2, 1, nullptr, 1,
                                             A2, 3 * HID, T1, HID, T2, HID);
  mixhop_spmm<<<spmmGrid, 256, 0, stream>>>(rowsI, colsI, vals, E, T1, HID, A2 + HID, 3 * HID);
  hipMemsetAsync(T1, 0, (size_t)N * HID * sizeof(float), stream);
  mixhop_spmm<<<spmmGrid, 256, 0, stream>>>(rowsI, colsI, vals, E, T2, HID, T1, HID);
  mixhop_spmm<<<spmmGrid, 256, 0, stream>>>(rowsI, colsI, vals, E, T1, HID, A2 + 2 * HID, 3 * HID);
  {
    long total = (long)N * 3 * HID;
    int  g = (int)((total + 255) / 256);
    mixhop_bias600<<<g, 256, 0, stream>>>(A2, b2, total);
  }

  // ================= Layer 3: A2 @ Wfc + bfc, log_softmax ========================
  mixhop_wmma_gemm<<<g64, blk, 0, stream>>>(A2, 3 * HID, N, 3 * HID, CLS, Wfc, 0, bfc, 2,
                                            node_emb, CLS, node_emb, CLS, node_emb, CLS);
  {
    int g = (N * 32 + 255) / 256;
    mixhop_logsoftmax<<<g, 256, 0, stream>>>(node_emb, preds, N);
  }
}
